// MetaGraphMeanLayer_38757784879145
// MI455X (gfx1250) — matrix operand order repaired and checked by reading, not compiled, not run on hardware
//
#include <hip/hip_runtime.h>

// MetaLayer GNN (edge MLP + node MLP + scatter-mean) fused for gfx1250.
// bf16 WMMA (v_wmma_f32_16x16x32_bf16) with f32 accumulation.
// Each wave owns 32 edges (two 16-row M-tiles) so every B fragment load
// feeds two WMMAs -> halves L2 weight traffic vs one tile per wave.

#define NN  50000
#define NE  600000
#define CHN 128

typedef __attribute__((ext_vector_type(16))) __bf16 v16bf;
typedef __attribute__((ext_vector_type(8)))  __bf16 v8bf;
typedef __attribute__((ext_vector_type(8)))  float  v8f;

union FragU { v16bf v; v8bf h[2]; unsigned short u[16]; };

// round-to-nearest-even f32 -> bf16 (bit pattern)
static __device__ __forceinline__ unsigned short f2bf(float f) {
  unsigned int u = __float_as_uint(f);
  unsigned int r = (u + 0x7FFFu + ((u >> 16) & 1u)) >> 16;
  return (unsigned short)r;
}

// A-matrix 16x32 bf16 fragment (ISA 7.12.2): lane<16 holds row M=lane,
// K = {khalf*8..+7, 16+khalf*8..+7} within the 32-wide K tile (khalf=lane>>4).
static __device__ __forceinline__ v16bf load_a_bf(const unsigned short* rowp, int f, int khalf) {
  FragU x;
  const unsigned short* p0 = rowp + f * 32 + khalf * 8;
  x.h[0] = *(const v8bf*)(p0);
  x.h[1] = *(const v8bf*)(p0 + 16);
  return x.v;
}

static __device__ __forceinline__ v16bf load_a_f32(const float* rowp, int f, int khalf) {
  FragU x;
  const float* p0 = rowp + f * 32 + khalf * 8;
#pragma unroll
  for (int i = 0; i < 8; ++i) x.u[i] = f2bf(p0[i]);
#pragma unroll
  for (int i = 0; i < 8; ++i) x.u[8 + i] = f2bf(p0[16 + i]);
  return x.v;
}

// B-matrix 32x16 bf16 fragment, pre-packed so each lane reads 32 contiguous bytes.
static __device__ __forceinline__ v16bf load_b(const unsigned short* wbase, int tile, int lane) {
  FragU x;
  const unsigned short* p = wbase + tile * 512 + lane * 16;
  x.h[0] = *(const v8bf*)p;
  x.h[1] = *(const v8bf*)(p + 8);
  return x.v;
}

#define WMMA_BF16(A, B, C) \
  __builtin_amdgcn_wmma_f32_16x16x32_bf16(false, (A), false, (B), (short)0, (C), false, false)

// ---- prep kernels -----------------------------------------------------------

__global__ void cvt_x_kernel(const float* __restrict__ x, unsigned short* __restrict__ xbf, int n) {
  int i = blockIdx.x * blockDim.x + threadIdx.x;
  if (i < n) xbf[i] = f2bf(x[i]);
}

// Pack W [K, 128] fp32 into B-fragment order: tile t = ktile*8 + ntile, lane
// holds N = ntile*16 + (lane&15), K = ktile*32 + (lane>>4)*8 + j + (j>=8 ? 8 : 0) (j=0..15).
__global__ void pack_w_kernel(const float* __restrict__ W, int ktiles, unsigned short* __restrict__ dst) {
  int gid = blockIdx.x * blockDim.x + threadIdx.x;
  if (gid >= ktiles * 8 * 32) return;
  int lane = gid & 31;
  int t = gid >> 5;
  int kt = t >> 3, nt = t & 7;
  int kbase = kt * 32 + (lane >> 4) * 8;
  int n = nt * 16 + (lane & 15);
  unsigned short* o = dst + t * 512 + lane * 16;
#pragma unroll
  for (int j = 0; j < 16; ++j) o[j] = f2bf(W[(size_t)(kbase + j + (j >= 8 ? 8 : 0)) * CHN + n]);
}

__global__ void finalize_kernel(float* __restrict__ newx, const float* __restrict__ cnt, int total) {
  int i = blockIdx.x * blockDim.x + threadIdx.x;
  if (i < total) newx[i] = newx[i] / fmaxf(cnt[i >> 7], 1.0f);
}

// ---- fused edge+node MLP kernel --------------------------------------------
// 1 wave = 32 edges (two M-tiles). 2 waves/block. 18750 jobs -> 9375 blocks.

__global__ __launch_bounds__(64) void fused_kernel(
    const unsigned short* __restrict__ xbf,
    const int* __restrict__ eidx,                 // [2*NE]
    const float* __restrict__ eattr,              // [NE, 128]
    const float* __restrict__ eb1, const float* __restrict__ eb2,
    const float* __restrict__ nb1, const float* __restrict__ nb2,
    const unsigned short* __restrict__ wE1, const unsigned short* __restrict__ wE2,
    const unsigned short* __restrict__ wN1, const unsigned short* __restrict__ wN2,
    float* __restrict__ newx, float* __restrict__ newe, float* __restrict__ cnt) {
  const int lane  = threadIdx.x & 31;
  const int w     = threadIdx.x >> 5;
  const int khalf = lane >> 4;
  const int mrow  = lane & 15;
  const int e0    = (blockIdx.x * 2 + w) * 32;    // first edge of tile 0
  const int eM0   = e0 + mrow;
  const int eM1   = e0 + 16 + mrow;

  // hidden activations round-trip (D-layout store -> A-layout reload); +8 pad vs banks
  __shared__ __align__(16) unsigned short hid[2][2][16][CHN + 8];

  const int r0 = eidx[eM0], c0 = eidx[NE + eM0];
  const int r1 = eidx[eM1], c1 = eidx[NE + eM1];

  int rows0[8], rows1[8];          // row index of edge for M = g + 8*khalf
#pragma unroll
  for (int g = 0; g < 8; ++g) rows0[g] = __shfl(r0, g + 8 * khalf, 32);
#pragma unroll
  for (int g = 0; g < 8; ++g) rows1[g] = __shfl(r1, g + 8 * khalf, 32);

  // A fragments of e_in = [x[row] | x[col] | edge_attr] : 16 x 384 bf16, x2 tiles
  v16bf aX0[12], aX1[12];
  {
    const unsigned short* xr0 = xbf + (size_t)r0 * CHN;
    const unsigned short* xc0 = xbf + (size_t)c0 * CHN;
    const unsigned short* xr1 = xbf + (size_t)r1 * CHN;
    const unsigned short* xc1 = xbf + (size_t)c1 * CHN;
    const float* ea0 = eattr + (size_t)eM0 * CHN;
    const float* ea1 = eattr + (size_t)eM1 * CHN;
#pragma unroll
    for (int f = 0; f < 4; ++f) { aX0[f]     = load_a_bf(xr0, f, khalf);
                                  aX1[f]     = load_a_bf(xr1, f, khalf); }
#pragma unroll
    for (int f = 0; f < 4; ++f) { aX0[4 + f] = load_a_bf(xc0, f, khalf);
                                  aX1[4 + f] = load_a_bf(xc1, f, khalf); }
#pragma unroll
    for (int f = 0; f < 4; ++f) { aX0[8 + f] = load_a_f32(ea0, f, khalf);
                                  aX1[8 + f] = load_a_f32(ea1, f, khalf); }
  }
  const unsigned short* hrow0 = &hid[w][0][mrow][0];
  const unsigned short* hrow1 = &hid[w][1][mrow][0];

  // ---- edge MLP layer 1: relu(e_in @ ew1 + eb1) -> hid (bf16)
#pragma unroll 1
  for (int n = 0; n < 8; ++n) {
    const float b = eb1[n * 16 + mrow];           // bias depends only on N
    v8f a0 = {b, b, b, b, b, b, b, b};
    v8f a1 = {b, b, b, b, b, b, b, b};
#pragma unroll
    for (int k = 0; k < 12; ++k) {
      const v16bf vb = load_b(wE1, k * 8 + n, lane);
      a0 = WMMA_BF16(aX0[k], vb, a0);
      a1 = WMMA_BF16(aX1[k], vb, a1);
    }
#pragma unroll
    for (int g = 0; g < 8; ++g) {                 // D layout: M = g + 8*(lane/16)
      hid[w][0][g + 8 * khalf][n * 16 + mrow] = f2bf(fmaxf(a0[g], 0.f));
      hid[w][1][g + 8 * khalf][n * 16 + mrow] = f2bf(fmaxf(a1[g], 0.f));
    }
  }
  v16bf hA0[4], hA1[4];
#pragma unroll
  for (int f = 0; f < 4; ++f) { hA0[f] = load_a_bf(hrow0, f, khalf);
                                hA1[f] = load_a_bf(hrow1, f, khalf); }

  // ---- edge MLP layer 2 -> new_edge_attr (f32 out) + bf16 copy back to hid
#pragma unroll 1
  for (int n = 0; n < 8; ++n) {
    const float b = eb2[n * 16 + mrow];
    v8f a0 = {b, b, b, b, b, b, b, b};
    v8f a1 = {b, b, b, b, b, b, b, b};
#pragma unroll
    for (int k = 0; k < 4; ++k) {
      const v16bf vb = load_b(wE2, k * 8 + n, lane);
      a0 = WMMA_BF16(hA0[k], vb, a0);
      a1 = WMMA_BF16(hA1[k], vb, a1);
    }
#pragma unroll
    for (int g = 0; g < 8; ++g) {
      const int M = g + 8 * khalf;
      newe[(size_t)(e0 + M) * CHN + n * 16 + mrow]      = a0[g];
      newe[(size_t)(e0 + 16 + M) * CHN + n * 16 + mrow] = a1[g];
      hid[w][0][M][n * 16 + mrow] = f2bf(a0[g]);
      hid[w][1][M][n * 16 + mrow] = f2bf(a1[g]);
    }
  }
  v16bf eA0[4], eA1[4];
#pragma unroll
  for (int f = 0; f < 4; ++f) { eA0[f] = load_a_bf(hrow0, f, khalf);
                                eA1[f] = load_a_bf(hrow1, f, khalf); }

  // ---- node MLP layer 1: relu([x[col] | new_edge_attr] @ nw1 + nb1)
#pragma unroll 1
  for (int n = 0; n < 8; ++n) {
    const float b = nb1[n * 16 + mrow];
    v8f a0 = {b, b, b, b, b, b, b, b};
    v8f a1 = {b, b, b, b, b, b, b, b};
#pragma unroll
    for (int k = 0; k < 4; ++k) {                 // x[col] part (aX[4..7])
      const v16bf vb = load_b(wN1, k * 8 + n, lane);
      a0 = WMMA_BF16(aX0[4 + k], vb, a0);
      a1 = WMMA_BF16(aX1[4 + k], vb, a1);
    }
#pragma unroll
    for (int k = 0; k < 4; ++k) {                 // new_edge_attr part
      const v16bf vb = load_b(wN1, (4 + k) * 8 + n, lane);
      a0 = WMMA_BF16(eA0[k], vb, a0);
      a1 = WMMA_BF16(eA1[k], vb, a1);
    }
#pragma unroll
    for (int g = 0; g < 8; ++g) {
      hid[w][0][g + 8 * khalf][n * 16 + mrow] = f2bf(fmaxf(a0[g], 0.f));
      hid[w][1][g + 8 * khalf][n * 16 + mrow] = f2bf(fmaxf(a1[g], 0.f));
    }
  }
  v16bf h20[4], h21[4];
#pragma unroll
  for (int f = 0; f < 4; ++f) { h20[f] = load_a_bf(hrow0, f, khalf);
                                h21[f] = load_a_bf(hrow1, f, khalf); }

  // ---- node MLP layer 2 -> msg, scatter-add into new_x by row
#pragma unroll 1
  for (int n = 0; n < 8; ++n) {
    const float b = nb2[n * 16 + mrow];
    v8f a0 = {b, b, b, b, b, b, b, b};
    v8f a1 = {b, b, b, b, b, b, b, b};
#pragma unroll
    for (int k = 0; k < 4; ++k) {
      const v16bf vb = load_b(wN2, k * 8 + n, lane);
      a0 = WMMA_BF16(h20[k], vb, a0);
      a1 = WMMA_BF16(h21[k], vb, a1);
    }
#pragma unroll
    for (int g = 0; g < 8; ++g) {
      atomicAdd(&newx[(size_t)rows0[g] * CHN + n * 16 + mrow], a0[g]);
      atomicAdd(&newx[(size_t)rows1[g] * CHN + n * 16 + mrow], a1[g]);
    }
  }
  if (lane < 16) { atomicAdd(&cnt[r0], 1.0f); atomicAdd(&cnt[r1], 1.0f); }
}

// ---- host launcher ----------------------------------------------------------

extern "C" void kernel_launch(void* const* d_in, const int* in_sizes, int n_in,
                              void* d_out, int out_size, void* d_ws, size_t ws_size,
                              hipStream_t stream) {
  (void)in_sizes; (void)n_in; (void)out_size; (void)ws_size;
  const float* x     = (const float*)d_in[0];
  const int*   eidx  = (const int*)d_in[1];      // int32 (x64 disabled)
  const float* eattr = (const float*)d_in[2];
  const float* ew1   = (const float*)d_in[3];
  const float* eb1   = (const float*)d_in[4];
  const float* ew2   = (const float*)d_in[5];
  const float* eb2   = (const float*)d_in[6];
  const float* nw1   = (const float*)d_in[7];
  const float* nb1   = (const float*)d_in[8];
  const float* nw2   = (const float*)d_in[9];
  const float* nb2   = (const float*)d_in[10];

  float* newx = (float*)d_out;                    // [NN, 128]
  float* newe = newx + (size_t)NN * CHN;          // [NE, 128]

  // workspace layout (~12.6 MB)
  char* ws = (char*)d_ws;
  unsigned short* xbf = (unsigned short*)ws;                         // 12,800,000 B
  float*          cnt = (float*)(ws + (size_t)NN * CHN * 2);         //    200,000 B
  unsigned short* wE1 = (unsigned short*)(ws + 13000000);            //     98,304 B (96 tiles)
  unsigned short* wE2 = (unsigned short*)(ws + 13098304);            //     32,768 B (32 tiles)
  unsigned short* wN1 = (unsigned short*)(ws + 13131072);            //     65,536 B (64 tiles)
  unsigned short* wN2 = (unsigned short*)(ws + 13196608);            //     32,768 B (32 tiles)

  hipMemsetAsync(newx, 0, (size_t)NN * CHN * sizeof(float), stream);
  hipMemsetAsync(cnt, 0, (size_t)NN * sizeof(float), stream);

  cvt_x_kernel<<<(NN * CHN + 255) / 256, 256, 0, stream>>>(x, xbf, NN * CHN);
  pack_w_kernel<<<(96 * 32 + 255) / 256, 256, 0, stream>>>(ew1, 12, wE1);
  pack_w_kernel<<<(32 * 32 + 255) / 256, 256, 0, stream>>>(ew2, 4, wE2);
  pack_w_kernel<<<(64 * 32 + 255) / 256, 256, 0, stream>>>(nw1, 8, wN1);
  pack_w_kernel<<<(32 * 32 + 255) / 256, 256, 0, stream>>>(nw2, 4, wN2);

  fused_kernel<<<NE / 64, 64, 0, stream>>>(xbf, eidx, eattr, eb1, eb2, nb1, nb2,
                                           wE1, wE2, wN1, wN2, newx, newe, cnt);

  finalize_kernel<<<(NN * CHN + 255) / 256, 256, 0, stream>>>(newx, cnt, NN * CHN);
}
